// SparseAttention_20641612824807
// MI455X (gfx1250) — matrix operand order repaired and checked by reading, not compiled, not run on hardware
//
#include <hip/hip_runtime.h>
#include <hip/hip_bf16.h>

// ---------------------------------------------------------------------------
// Sparse (Longformer-style) attention for MI455X / gfx1250.
// bf16 WMMA (v_wmma_f32_16x16x32_bf16) for all three GEMM stages.
// GEMM wave tile widened to 32x64 (8 accumulators) so each B fragment is
// reused by two WMMAs: 12 b128 loads per 8 WMMAs per k-chunk.
// ---------------------------------------------------------------------------

#define S_LEN  4096
#define DMODEL 1024
#define NHEAD  16
#define HDIM   64
#define QKVN   (3 * NHEAD * HDIM)   // 3072

typedef __attribute__((ext_vector_type(16))) __bf16 v16bf;
typedef __attribute__((ext_vector_type(8)))  float  v8f;

union Frag16 {
    v16bf v;
    float4 f4[2];
    unsigned short u16[16];
};

__device__ __forceinline__ unsigned short f32_to_bf16_rne(float f) {
    unsigned int u = __float_as_uint(f);
    unsigned int r = u + 0x7FFFu + ((u >> 16) & 1u);
    return (unsigned short)(r >> 16);
}

__device__ __forceinline__ v8f wmma_bf16(const Frag16& a, const Frag16& b, v8f c) {
    return __builtin_amdgcn_wmma_f32_16x16x32_bf16(
        /*neg_a=*/false, a.v, /*neg_b=*/false, b.v,
        /*c_mod=*/(short)0, c, /*reuse_a=*/false, /*reuse_b=*/false);
}

// ---------------------------------------------------------------------------
// Elementwise f32 -> bf16
// ---------------------------------------------------------------------------
__global__ __launch_bounds__(256) void cvt_f32_to_bf16(
    const float* __restrict__ src, unsigned short* __restrict__ dst, int n) {
    int i = blockIdx.x * blockDim.x + threadIdx.x;
    if (i < n) dst[i] = f32_to_bf16_rne(src[i]);
}

// src [K][N] f32 row-major -> dst [N][K] bf16 row-major (transpose+convert)
__global__ __launch_bounds__(256) void cvt_transpose_bf16(
    const float* __restrict__ src, unsigned short* __restrict__ dst, int K, int N) {
    int idx = blockIdx.x * blockDim.x + threadIdx.x;
    if (idx < K * N) {
        int col = idx / K;   // output row == original column
        int k   = idx % K;
        dst[idx] = f32_to_bf16_rne(src[k * N + col]);
    }
}

// ---------------------------------------------------------------------------
// QKV projection: [4096,1024]bf16 @ [1024,3072] -> scatter into per-head
// Q/K/V buffers, layout [H][S][64] bf16. W pre-transposed: WqkvT[3072][1024].
// One wave computes a 32x64 output tile (2 M-subtiles x 4 N-subtiles).
// ---------------------------------------------------------------------------
__global__ __launch_bounds__(256) void qkv_gemm(
    const unsigned short* __restrict__ Xb,     // [4096][1024]
    const unsigned short* __restrict__ WqkvT,  // [3072][1024]
    unsigned short* __restrict__ Qb,           // [16][4096][64]
    unsigned short* __restrict__ Kb,
    unsigned short* __restrict__ Vb) {
    const int lane  = threadIdx.x & 31;
    const int wave  = threadIdx.x >> 5;                 // 0..7
    const int lrow  = lane & 15;
    const int lhalf = lane >> 4;
    const int m0 = (blockIdx.x * 8 + wave) * 32;        // row of 4096
    const int n0 = blockIdx.y * 64;                     // col of 3072

    v8f acc[2][4] = {{v8f{}, v8f{}, v8f{}, v8f{}},
                     {v8f{}, v8f{}, v8f{}, v8f{}}};

    const unsigned short* arow0 = Xb + (size_t)(m0 + lrow) * DMODEL + lhalf * 8;
    const unsigned short* arow1 = arow0 + (size_t)16 * DMODEL;
    for (int k0 = 0; k0 < DMODEL; k0 += 32) {
        Frag16 a0, a1;
        a0.f4[0] = *(const float4*)(arow0 + k0);
        a0.f4[1] = *(const float4*)(arow0 + k0 + 16);
        a1.f4[0] = *(const float4*)(arow1 + k0);
        a1.f4[1] = *(const float4*)(arow1 + k0 + 16);
        __builtin_prefetch(arow0 + k0 + 64, 0, 3);
        __builtin_prefetch(arow1 + k0 + 64, 0, 3);
#pragma unroll
        for (int g = 0; g < 4; ++g) {
            const unsigned short* brow =
                WqkvT + (size_t)(n0 + g * 16 + lrow) * DMODEL + k0 + lhalf * 8;
            Frag16 b;
            b.f4[0] = *(const float4*)(brow);
            b.f4[1] = *(const float4*)(brow + 16);
            acc[0][g] = wmma_bf16(a0, b, acc[0][g]);
            acc[1][g] = wmma_bf16(a1, b, acc[1][g]);
        }
    }

    // scatter: column c -> (t = q/k/v, head h, dim d)
#pragma unroll
    for (int g = 0; g < 4; ++g) {
        int c  = n0 + g * 16 + lrow;           // lane holds column lrow of tile
        int t  = c / (NHEAD * HDIM);
        int hd = c % (NHEAD * HDIM);
        int h = hd / HDIM, d = hd % HDIM;
        unsigned short* dst =
            (t == 0 ? Qb : (t == 1 ? Kb : Vb)) + (size_t)h * S_LEN * HDIM + d;
#pragma unroll
        for (int mt = 0; mt < 2; ++mt) {
#pragma unroll
            for (int r = 0; r < 8; ++r) {
                int row = m0 + mt * 16 + r + 8 * lhalf;   // C layout
                dst[(size_t)row * HDIM] = f32_to_bf16_rne(acc[mt][g][r]);
            }
        }
    }
}

// ---------------------------------------------------------------------------
// Sparse attention, flash-style, one wave per (head, 16-query tile).
// Keys processed in blocks of 32: global blocks [0,64) then the causal
// local band [max(64, align32(i0-128)), align32(i0+16)).
// ---------------------------------------------------------------------------
__global__ __launch_bounds__(32) void sparse_attn(
    const unsigned short* __restrict__ Qb,
    const unsigned short* __restrict__ Kb,
    const unsigned short* __restrict__ Vb,
    unsigned short* __restrict__ Ob) {         // [4096][1024] bf16
    __shared__ unsigned short vt[HDIM * 32];   // V^T tile: vt[d][k]
    __shared__ unsigned short pl[16 * 32];     // P tile:   pl[m][k]

    const int lane  = threadIdx.x;
    const int lrow  = lane & 15;
    const int lhalf = lane >> 4;
    const int h  = blockIdx.y;
    const int i0 = blockIdx.x * 16;

    const unsigned short* Qh = Qb + (size_t)h * S_LEN * HDIM;
    const unsigned short* Kh = Kb + (size_t)h * S_LEN * HDIM;
    const unsigned short* Vh = Vb + (size_t)h * S_LEN * HDIM;

    // Q fragments for the two 32-wide K chunks of the head dim
    Frag16 qf[2];
    {
        const unsigned short* qrow = Qh + (size_t)(i0 + lrow) * HDIM + lhalf * 16;
        qf[0].f4[0] = *(const float4*)(qrow);
        qf[0].f4[1] = *(const float4*)(qrow + 8);
        qf[1].f4[0] = *(const float4*)(qrow + 32);
        qf[1].f4[1] = *(const float4*)(qrow + 40);
    }

    v8f o[4] = {v8f{}, v8f{}, v8f{}, v8f{}};
    float mrow[8], lsum[8];
#pragma unroll
    for (int r = 0; r < 8; ++r) { mrow[r] = -1e30f; lsum[r] = 0.0f; }

    const float scale = 0.125f;  // 1/sqrt(64)

    int jloc0 = (i0 - 128) & ~31;
    if (jloc0 < 64) jloc0 = 64;
    int jend = (i0 + 16 + 31) & ~31;

    for (int phase = 0; phase < 2; ++phase) {
        const int jb_lo = (phase == 0) ? 0  : jloc0;
        const int jb_hi = (phase == 0) ? 64 : jend;
        for (int jb = jb_lo; jb < jb_hi; jb += 32) {
            __syncthreads();  // single-wave WG: near-free; protects LDS reuse

            // ---- scores: two 16x16 tiles over 32 keys --------------------
            v8f s0 = v8f{}, s1 = v8f{};
#pragma unroll
            for (int kc = 0; kc < 2; ++kc) {
                Frag16 b0, b1;
                const unsigned short* kr0 =
                    Kh + (size_t)(jb + lrow) * HDIM + kc * 32 + lhalf * 16;
                const unsigned short* kr1 =
                    Kh + (size_t)(jb + 16 + lrow) * HDIM + kc * 32 + lhalf * 16;
                b0.f4[0] = *(const float4*)(kr0);
                b0.f4[1] = *(const float4*)(kr0 + 8);
                b1.f4[0] = *(const float4*)(kr1);
                b1.f4[1] = *(const float4*)(kr1 + 8);
                s0 = wmma_bf16(qf[kc], b0, s0);
                s1 = wmma_bf16(qf[kc], b1, s1);
            }

            // ---- stage V^T into LDS (each lane owns one key row) ---------
            {
                union { float4 f4[8]; unsigned short s[64]; } vv;
                const unsigned short* vrowp = Vh + (size_t)(jb + lane) * HDIM;
#pragma unroll
                for (int q = 0; q < 8; ++q) vv.f4[q] = *(const float4*)(vrowp + q * 8);
#pragma unroll
                for (int d = 0; d < 64; ++d) vt[d * 32 + lane] = vv.s[d];
            }

            // ---- mask + online softmax (C layout) ------------------------
            float p0[8], p1[8], bmax[8];
            const int j0c = jb + lrow;        // column of s0 held by this lane
            const int j1c = jb + 16 + lrow;   // column of s1
#pragma unroll
            for (int r = 0; r < 8; ++r) {
                int i = i0 + r + 8 * lhalf;
                float v0 = s0[r] * scale;
                float v1 = s1[r] * scale;
                bool ok0 = (j0c <= i) && ((j0c >= i - 128) || (j0c < 64) || (i < 64));
                bool ok1 = (j1c <= i) && ((j1c >= i - 128) || (j1c < 64) || (i < 64));
                v0 = ok0 ? v0 : -1e30f;
                v1 = ok1 ? v1 : -1e30f;
                p0[r] = v0; p1[r] = v1;
                float mx = fmaxf(v0, v1);
                mx = fmaxf(mx, __shfl_xor(mx, 1));
                mx = fmaxf(mx, __shfl_xor(mx, 2));
                mx = fmaxf(mx, __shfl_xor(mx, 4));
                mx = fmaxf(mx, __shfl_xor(mx, 8));
                bmax[r] = mx;
            }
#pragma unroll
            for (int r = 0; r < 8; ++r) {
                float mnew  = fmaxf(mrow[r], bmax[r]);
                float alpha = __expf(mrow[r] - mnew);
                mrow[r] = mnew;
                float e0 = __expf(p0[r] - mnew);
                float e1 = __expf(p1[r] - mnew);
                p0[r] = e0; p1[r] = e1;
                float rs = e0 + e1;
                rs += __shfl_xor(rs, 1);
                rs += __shfl_xor(rs, 2);
                rs += __shfl_xor(rs, 4);
                rs += __shfl_xor(rs, 8);
                lsum[r] = lsum[r] * alpha + rs;
                o[0][r] *= alpha; o[1][r] *= alpha;
                o[2][r] *= alpha; o[3][r] *= alpha;
            }

            // ---- P: C layout -> LDS -> A layout --------------------------
#pragma unroll
            for (int r = 0; r < 8; ++r) {
                int row = r + 8 * lhalf;
                pl[row * 32 + lrow]      = f32_to_bf16_rne(p0[r]);
                pl[row * 32 + 16 + lrow] = f32_to_bf16_rne(p1[r]);
            }
            __syncthreads();

            Frag16 pa;
            {
                const unsigned short* prow = pl + lrow * 32 + lhalf * 8;
                pa.f4[0] = *(const float4*)(prow);
                pa.f4[1] = *(const float4*)(prow + 16);
            }
            // ---- O += P @ V (4 N-groups of 16 dims) ----------------------
#pragma unroll
            for (int g = 0; g < 4; ++g) {
                Frag16 bv;
                const unsigned short* vrow = vt + (g * 16 + lrow) * 32 + lhalf * 8;
                bv.f4[0] = *(const float4*)(vrow);
                bv.f4[1] = *(const float4*)(vrow + 16);
                o[g] = wmma_bf16(pa, bv, o[g]);
            }
        }
    }

    // ---- normalize + write [S][H*64] bf16 --------------------------------
#pragma unroll
    for (int r = 0; r < 8; ++r) {
        float inv = 1.0f / lsum[r];
        int row = i0 + r + 8 * lhalf;
        unsigned short* orow =
            Ob + (size_t)row * (NHEAD * HDIM) + h * HDIM + lrow;
        orow[0]  = f32_to_bf16_rne(o[0][r] * inv);
        orow[16] = f32_to_bf16_rne(o[1][r] * inv);
        orow[32] = f32_to_bf16_rne(o[2][r] * inv);
        orow[48] = f32_to_bf16_rne(o[3][r] * inv);
    }
}

// ---------------------------------------------------------------------------
// Output projection: [4096,1024]bf16 @ [1024,1024] -> f32 out.
// WoutT pre-transposed to [N][K]. 32x64 wave tile.
// ---------------------------------------------------------------------------
__global__ __launch_bounds__(256) void out_gemm(
    const unsigned short* __restrict__ Ab,     // [4096][1024] attn (bf16)
    const unsigned short* __restrict__ WoutT,  // [1024][1024]
    float* __restrict__ out) {                 // [4096][1024] f32
    const int lane  = threadIdx.x & 31;
    const int wave  = threadIdx.x >> 5;
    const int lrow  = lane & 15;
    const int lhalf = lane >> 4;
    const int m0 = (blockIdx.x * 8 + wave) * 32;
    const int n0 = blockIdx.y * 64;

    v8f acc[2][4] = {{v8f{}, v8f{}, v8f{}, v8f{}},
                     {v8f{}, v8f{}, v8f{}, v8f{}}};

    const unsigned short* arow0 = Ab + (size_t)(m0 + lrow) * DMODEL + lhalf * 8;
    const unsigned short* arow1 = arow0 + (size_t)16 * DMODEL;
    for (int k0 = 0; k0 < DMODEL; k0 += 32) {
        Frag16 a0, a1;
        a0.f4[0] = *(const float4*)(arow0 + k0);
        a0.f4[1] = *(const float4*)(arow0 + k0 + 16);
        a1.f4[0] = *(const float4*)(arow1 + k0);
        a1.f4[1] = *(const float4*)(arow1 + k0 + 16);
        __builtin_prefetch(arow0 + k0 + 64, 0, 3);
        __builtin_prefetch(arow1 + k0 + 64, 0, 3);
#pragma unroll
        for (int g = 0; g < 4; ++g) {
            const unsigned short* brow =
                WoutT + (size_t)(n0 + g * 16 + lrow) * DMODEL + k0 + lhalf * 8;
            Frag16 b;
            b.f4[0] = *(const float4*)(brow);
            b.f4[1] = *(const float4*)(brow + 16);
            acc[0][g] = wmma_bf16(a0, b, acc[0][g]);
            acc[1][g] = wmma_bf16(a1, b, acc[1][g]);
        }
    }
#pragma unroll
    for (int g = 0; g < 4; ++g) {
        int c = n0 + g * 16 + lrow;
#pragma unroll
        for (int mt = 0; mt < 2; ++mt) {
#pragma unroll
            for (int r = 0; r < 8; ++r) {
                int row = m0 + mt * 16 + r + 8 * lhalf;
                out[(size_t)row * DMODEL + c] = acc[mt][g][r];
            }
        }
    }
}

// ---------------------------------------------------------------------------
extern "C" void kernel_launch(void* const* d_in, const int* in_sizes, int n_in,
                              void* d_out, int out_size, void* d_ws, size_t ws_size,
                              hipStream_t stream) {
    const float* x     = (const float*)d_in[0];  // [1,4096,1024]
    const float* w_qkv = (const float*)d_in[1];  // [1024,3072]
    const float* w_out = (const float*)d_in[2];  // [1024,1024]
    float* out = (float*)d_out;                  // [1,4096,1024]

    // workspace partition (bf16 buffers), 256B aligned slices
    size_t off = 0;
    auto take = [&](size_t bytes) {
        void* p = (char*)d_ws + off;
        off += (bytes + 255) & ~(size_t)255;
        return p;
    };
    unsigned short* Xb    = (unsigned short*)take((size_t)S_LEN * DMODEL * 2);  // 8 MB
    unsigned short* WqkvT = (unsigned short*)take((size_t)QKVN * DMODEL * 2);   // 6 MB
    unsigned short* WoutT = (unsigned short*)take((size_t)DMODEL * DMODEL * 2); // 2 MB
    unsigned short* Qb    = (unsigned short*)take((size_t)NHEAD * S_LEN * HDIM * 2);
    unsigned short* Kb    = (unsigned short*)take((size_t)NHEAD * S_LEN * HDIM * 2);
    unsigned short* Vb    = (unsigned short*)take((size_t)NHEAD * S_LEN * HDIM * 2);
    unsigned short* Ob    = (unsigned short*)take((size_t)S_LEN * DMODEL * 2);  // 8 MB

    {   // conversions
        int n = S_LEN * DMODEL;
        cvt_f32_to_bf16<<<(n + 255) / 256, 256, 0, stream>>>(x, Xb, n);
        int nq = DMODEL * QKVN;
        cvt_transpose_bf16<<<(nq + 255) / 256, 256, 0, stream>>>(w_qkv, WqkvT, DMODEL, QKVN);
        int no = DMODEL * DMODEL;
        cvt_transpose_bf16<<<(no + 255) / 256, 256, 0, stream>>>(w_out, WoutT, DMODEL, DMODEL);
    }

    // QKV projection: 4096x3072, wave tile 32x64, block = 8 waves over M
    qkv_gemm<<<dim3(S_LEN / 256, QKVN / 64), 256, 0, stream>>>(Xb, WqkvT, Qb, Kb, Vb);

    // sparse attention: one wave per (16-query tile, head)
    sparse_attn<<<dim3(S_LEN / 16, NHEAD), 32, 0, stream>>>(Qb, Kb, Vb, Ob);

    // output projection: 4096x1024 f32
    out_gemm<<<dim3(S_LEN / 256, DMODEL / 64), 256, 0, stream>>>(Ob, WoutT, out);
}
